// MiniMaxText01LightningAttention_76562087018684
// MI455X (gfx1250) — matrix operand order repaired and checked by reading, not compiled, not run on hardware
//
#include <hip/hip_runtime.h>
#include <hip/hip_bf16.h>

typedef __attribute__((ext_vector_type(16))) _Float16 v16h;
typedef __attribute__((ext_vector_type(8)))  _Float16 v8h;
typedef __attribute__((ext_vector_type(4)))  _Float16 v4h;
typedef __attribute__((ext_vector_type(8)))  float    v8f;

#define BB    2
#define HH    64
#define SS    4096
#define DD    64
#define BLKR  256
#define NBLK  16
#define NTH   256
#define LDQ   72    // f16 row stride for qh/kh (64 + 8 pad -> conflict-free b128 column reads)
#define LDN   264   // f16 row stride for vT (256 + 8 pad)
#define KVLD  68    // f32 row stride for kv state (64 + 4 pad)
#define QKLD  40    // f16 row stride for per-wave qk scratch (32 + 8 pad)

static __device__ __forceinline__ v8f wmma16(v16h a, v16h b, v8f c) {
  // D = A(16x32 f16) * B(32x16 f16) + C(16x16 f32)
  return __builtin_amdgcn_wmma_f32_16x16x32_f16(false, a, false, b, (short)0, c, false, false);
}

static __device__ __forceinline__ v16h cat8(v8h lo, v8h hi) {
  v16h r;
#pragma unroll
  for (int i = 0; i < 8; ++i) { r[i] = lo[i]; r[8 + i] = hi[i]; }
  return r;
}

// A fragment (16x32, f16): lane r=lane%16 holds row M=r; a[0..7] = row[kb+half*8 .. +8),
// a[8..15] = row[kb+16+half*8 .. +8)   (ISA 7.12.2, 16-bit A 16x32)
static __device__ __forceinline__ v16h frag_a_row(const _Float16* rowp, int kb, int hl) {
  v8h lo = *(const v8h*)(rowp + kb + hl * 8);
  v8h hi = *(const v8h*)(rowp + kb + 16 + hl * 8);
  return cat8(lo, hi);
}

// B fragment (32x16, f16): lane holds col N=lane%16; b[0..7] = col[kb+half*8 .. +8),
// b[8..15] = col[kb+16+half*8 .. +8)
static __device__ __forceinline__ v16h frag_b_col(const _Float16* colp, int kb, int hl) {
  v8h lo = *(const v8h*)(colp + kb + hl * 8);
  v8h hi = *(const v8h*)(colp + kb + 16 + hl * 8);
  return cat8(lo, hi);
}

// Same B layout but sourced from f32 LDS (kv state), converting on the fly
static __device__ __forceinline__ v16h frag_b_col_f32(const float* colp, int kb, int hl) {
  v16h r;
  const float* p = colp + kb + hl * 8;
#pragma unroll
  for (int i = 0; i < 16; ++i) r[i] = (_Float16)p[(i < 8) ? i : (i + 8)];
  return r;
}

__global__ __launch_bounds__(NTH, 1)
void MiniMaxText01LightningAttention_76562087018684_kernel(
    const float* __restrict__ qg, const float* __restrict__ kg,
    const float* __restrict__ vg, const int* __restrict__ maskg,
    float* __restrict__ outg)
{
  __shared__ __align__(16) _Float16 qh[BLKR * LDQ];    // q block, [row][d] f16
  __shared__ __align__(16) _Float16 kh[BLKR * LDQ];    // k block, [row][d] f16
  __shared__ __align__(16) _Float16 vt[DD * LDN];      // v block transposed, [e][row] f16
  __shared__ __align__(16) _Float16 qks[8 * 16 * QKLD];// per-wave qk relayout scratch
  __shared__ __align__(16) float    kvT[DD * KVLD];    // recurrent state, kvT[e][d] = kv[d][e]
  __shared__ __align__(16) float    kdec[BLKR];        // k_decay per key row

  const int bh   = blockIdx.x;          // 0..127 == b*H + h
  const int b    = bh / HH;
  const int h    = bh % HH;
  const int tid  = threadIdx.x;
  const int lane = tid & 31;
  const int wave = tid >> 5;
  const int hl   = lane >> 4;           // lane half (0/1)
  const int lr   = lane & 15;

  // slopes[h] = 2^(-(h+1)/8) * (1 - L/(NL-1) + 1e-5)
  const float lscale = 1.0f - 5.0f / 31.0f + 1e-5f;
  const float slope  = exp2f(-(float)(h + 1) * 0.125f) * lscale;
  const float bdecay = __expf(-slope * (float)BLKR);

  const float* qb  = qg + (size_t)bh * SS * DD;
  const float* kgb = kg + (size_t)bh * SS * DD;
  const float* vbp = vg + (size_t)bh * SS * DD;
  const int*   mbp = maskg + (size_t)b * SS;
  float*       ob  = outg + (size_t)bh * SS * DD;

  for (int i = tid; i < BLKR; i += NTH) kdec[i] = __expf(-slope * (float)(BLKR - 1 - i));
  for (int i = tid; i < DD * KVLD; i += NTH) kvT[i] = 0.0f;

  const int m0 = wave * 32;             // this wave's 32-row strip of the block

  for (int blk = 0; blk < NBLK; ++blk) {
    __syncthreads();  // previous step fully consumed qh/kh/vt and wrote kvT

    // ---------------- load & convert q,k,v into LDS ----------------
    {
      const float4* qp = (const float4*)(qb  + (size_t)blk * BLKR * DD);
      const float4* kp = (const float4*)(kgb + (size_t)blk * BLKR * DD);
      const float4* vp = (const float4*)(vbp + (size_t)blk * BLKR * DD);
      const int*    mp = mbp + blk * BLKR;
#pragma unroll 4
      for (int idx = tid; idx < BLKR * DD / 4; idx += NTH) {
        int row = idx >> 4;
        int col = (idx & 15) << 2;
        float4 tq = qp[idx];
        float4 tk = kp[idx];
        float4 tv = vp[idx];
        v4h hq = { (_Float16)tq.x, (_Float16)tq.y, (_Float16)tq.z, (_Float16)tq.w };
        v4h hk = { (_Float16)tk.x, (_Float16)tk.y, (_Float16)tk.z, (_Float16)tk.w };
        *(v4h*)&qh[row * LDQ + col] = hq;
        *(v4h*)&kh[row * LDQ + col] = hk;
        float m = (float)mp[row];                     // mask padded positions in v
        vt[(col + 0) * LDN + row] = (_Float16)(tv.x * m);
        vt[(col + 1) * LDN + row] = (_Float16)(tv.y * m);
        vt[(col + 2) * LDN + row] = (_Float16)(tv.z * m);
        vt[(col + 3) * LDN + row] = (_Float16)(tv.w * m);
      }
      if (blk + 1 < NBLK) {  // pull next block into L2 while we compute
        const float* nq = qb  + (size_t)(blk + 1) * BLKR * DD;
        const float* nk = kgb + (size_t)(blk + 1) * BLKR * DD;
        const float* nv = vbp + (size_t)(blk + 1) * BLKR * DD;
        for (int l = tid; l < (BLKR * DD) / 32; l += NTH) {
          __builtin_prefetch(nq + (size_t)l * 32, 0, 0);
          __builtin_prefetch(nk + (size_t)l * 32, 0, 0);
          __builtin_prefetch(nv + (size_t)l * 32, 0, 0);
        }
      }
    }
    __syncthreads();

    // ---------------- per-wave q fragments (raw + q_decay-scaled) ----------------
    v16h qf[2][2], qsf[2][2];
#pragma unroll
    for (int mt = 0; mt < 2; ++mt) {
      const _Float16* qr = &qh[(m0 + mt * 16 + lr) * LDQ];
      _Float16 qdh = (_Float16)__expf(-slope * (float)(m0 + mt * 16 + lr + 1)); // per-row == per-lane
#pragma unroll
      for (int g = 0; g < 2; ++g) {
        v16h f = frag_a_row(qr, g * 32, hl);
        qf[mt][g] = f;
        v16h fs;
#pragma unroll
        for (int i = 0; i < 16; ++i) fs[i] = f[i] * qdh;
        qsf[mt][g] = fs;
      }
    }

    // ---------------- o_inter = (q * q_decay) @ kv_prev ----------------
    v8f oacc[2][4];
#pragma unroll
    for (int mt = 0; mt < 2; ++mt)
#pragma unroll
      for (int et = 0; et < 4; ++et) oacc[mt][et] = v8f{};
#pragma unroll
    for (int et = 0; et < 4; ++et) {
      const float* kvc = &kvT[(et * 16 + lr) * KVLD];   // column e, contiguous in d
      v16h b0 = frag_b_col_f32(kvc, 0,  hl);
      v16h b1 = frag_b_col_f32(kvc, 32, hl);
#pragma unroll
      for (int mt = 0; mt < 2; ++mt) {
        oacc[mt][et] = wmma16(qsf[mt][0], b0, oacc[mt][et]);
        oacc[mt][et] = wmma16(qsf[mt][1], b1, oacc[mt][et]);
      }
    }
    __syncthreads();  // every wave finished reading kvT; kv update may proceed later

    // ---------------- intra-block causal attention ----------------
    _Float16* myqks = &qks[wave * 16 * QKLD];
    for (int j = 0; j <= wave; ++j) {                   // triangular chunk loop (keys <= queries)
      int c0 = j * 32;
      const _Float16* kc0 = &kh[(c0 + lr) * LDQ];       // key column fragments (contig in d)
      const _Float16* kc1 = &kh[(c0 + 16 + lr) * LDQ];
      v16h kb00 = frag_b_col(kc0, 0,  hl);
      v16h kb01 = frag_b_col(kc0, 32, hl);
      v16h kb10 = frag_b_col(kc1, 0,  hl);
      v16h kb11 = frag_b_col(kc1, 32, hl);
      v16h vbf[4];
#pragma unroll
      for (int et = 0; et < 4; ++et) vbf[et] = frag_b_col(&vt[(et * 16 + lr) * LDN], c0, hl);
#pragma unroll
      for (int mt = 0; mt < 2; ++mt) {
        v8f qk0 = v8f{}, qk1 = v8f{};
        qk0 = wmma16(qf[mt][0], kb00, qk0);
        qk0 = wmma16(qf[mt][1], kb01, qk0);
        qk1 = wmma16(qf[mt][0], kb10, qk1);
        qk1 = wmma16(qf[mt][1], kb11, qk1);
        // causal decay on C-layout elements, spill to LDS to re-layout as A operand
        int mbase = m0 + mt * 16 + hl * 8;
#pragma unroll
        for (int vv = 0; vv < 8; ++vv) {
          int m  = mbase + vv;
          int n0 = c0 + lr;
          int n1 = c0 + 16 + lr;
          float d0 = (m >= n0) ? __expf(-slope * (float)(m - n0)) : 0.0f;
          float d1 = (m >= n1) ? __expf(-slope * (float)(m - n1)) : 0.0f;
          myqks[(hl * 8 + vv) * QKLD + lr]      = (_Float16)(qk0[vv] * d0);
          myqks[(hl * 8 + vv) * QKLD + 16 + lr] = (_Float16)(qk1[vv] * d1);
        }
        asm volatile("s_wait_dscnt 0" ::: "memory");    // LDS stores visible before re-read
        v16h aq = frag_a_row(&myqks[lr * QKLD], 0, hl);
#pragma unroll
        for (int et = 0; et < 4; ++et) oacc[mt][et] = wmma16(aq, vbf[et], oacc[mt][et]);
        asm volatile("s_wait_dscnt 0" ::: "memory");    // reads done before mt=1 overwrites scratch
      }
    }

    // ---------------- write output tile (o_inter + o_intra) ----------------
    float* op = ob + (size_t)blk * BLKR * DD;
#pragma unroll
    for (int mt = 0; mt < 2; ++mt) {
      int rbase = m0 + mt * 16 + hl * 8;
#pragma unroll
      for (int et = 0; et < 4; ++et)
#pragma unroll
        for (int vv = 0; vv < 8; ++vv)
          op[(size_t)(rbase + vv) * DD + et * 16 + lr] = oacc[mt][et][vv];
    }

    // ---------------- kv_new = bdecay*kv + (k * k_decay)^T @ v ----------------
    // 16 (d,e) tiles split 2-per-wave; each wave owns its tiles exclusively.
#pragma unroll
    for (int t = 0; t < 2; ++t) {
      int tile = wave * 2 + t;
      int dt = tile >> 2;
      int et = tile & 3;
      float* kvp = &kvT[(et * 16 + lr) * KVLD + dt * 16 + hl * 8];
      v8f acc;
#pragma unroll
      for (int vv = 0; vv < 8; ++vv) acc[vv] = kvp[vv] * bdecay;   // C = bdecay * kv_old
      const _Float16* vc = &vt[(et * 16 + lr) * LDN];
      int drow = dt * 16 + lr;                                      // A row M = d
#pragma unroll
      for (int c0 = 0; c0 < BLKR; c0 += 32) {
        v16h a;                                                     // A(m=d, k=n) = k[n][d]*kdec[n]
#pragma unroll
        for (int i = 0; i < 8; ++i) {
          int n0 = c0 + hl * 8 + i;
          int n1 = c0 + 16 + hl * 8 + i;
          a[i]     = (_Float16)((float)kh[n0 * LDQ + drow] * kdec[n0]);
          a[8 + i] = (_Float16)((float)kh[n1 * LDQ + drow] * kdec[n1]);
        }
        v16h bv = frag_b_col(vc, c0, hl);
        acc = wmma16(a, bv, acc);
      }
#pragma unroll
      for (int vv = 0; vv < 8; ++vv) kvp[vv] = acc[vv];             // write kv_new (own tile)
    }
  }
}

extern "C" void kernel_launch(void* const* d_in, const int* in_sizes, int n_in,
                              void* d_out, int out_size, void* d_ws, size_t ws_size,
                              hipStream_t stream) {
  (void)in_sizes; (void)n_in; (void)out_size; (void)d_ws; (void)ws_size;
  const float* q    = (const float*)d_in[0];
  const float* k    = (const float*)d_in[1];
  const float* v    = (const float*)d_in[2];
  const int*   mask = (const int*)d_in[3];
  float* out = (float*)d_out;
  dim3 grid(BB * HH), block(NTH);
  hipLaunchKernelGGL(MiniMaxText01LightningAttention_76562087018684_kernel,
                     grid, block, 0, stream, q, k, v, mask, out);
}
